// StyleBlock_31610959299065
// MI455X (gfx1250) — matrix operand order repaired and checked by reading, not compiled, not run on hardware
//
#include <hip/hip_runtime.h>
#include <hip/hip_bf16.h>

#define B_     8
#define CIN_   512
#define COUT_  512
#define SDIM_  512
#define SQRT2_ 1.41421356237309515f

typedef unsigned short ushort_t;
typedef __attribute__((ext_vector_type(16))) __bf16 v16bf;
typedef __attribute__((ext_vector_type(8)))  float  v8f;

__device__ __forceinline__ ushort_t f2bf(float f) {
  union { float f; unsigned u; } x; x.f = f;
  unsigned r = x.u + 0x7FFFu + ((x.u >> 16) & 1u);   // round-to-nearest-even
  return (ushort_t)(r >> 16);
}

// ---------------------------------------------------------------- style affine
// s[b][ci] = (style[b]*1/sqrt(S)) . aff_w[ci] + aff_b[ci] + 1
__global__ void style_kernel(const float* __restrict__ style,
                             const float* __restrict__ aff_w,
                             const float* __restrict__ aff_b,
                             float* __restrict__ s) {
  int gid = blockIdx.x * blockDim.x + threadIdx.x;   // B*CIN = 4096
  if (gid >= B_ * CIN_) return;
  int b = gid >> 9, ci = gid & 511;
  const float elr = 0.044194173824159216f;           // 1/sqrt(512)
  const float* st = style + (size_t)b * SDIM_;
  const float* aw = aff_w + (size_t)ci * SDIM_;
  float acc = 0.f;
  for (int k = 0; k < SDIM_; ++k) acc += st[k] * aw[k];
  s[gid] = acc * elr + aff_b[ci] + 1.0f;
}

// ---------------------------------------------------------------- demodulation
// d[b][co] = rsqrt( sum_{ci,k,k} (w*s*cs)^2 + 1e-4 )
__global__ void demod_kernel(const float* __restrict__ bw,   // [COUT][CIN][9]
                             const float* __restrict__ s,    // [B][CIN]
                             float* __restrict__ d) {
  __shared__ float red[256];
  int bco = blockIdx.x;                 // b*COUT+co
  int b = bco / COUT_, co = bco % COUT_;
  const float cs = 1.0f / 1536.0f;      // 1/sqrt(512*512*9)
  const float* w  = bw + (size_t)co * CIN_ * 9;
  const float* sb = s + (size_t)b * CIN_;
  float acc = 0.f;
  for (int j = threadIdx.x; j < CIN_ * 9; j += 256) {
    float v = w[j] * sb[j / 9] * cs;
    acc += v * v;
  }
  red[threadIdx.x] = acc;
  __syncthreads();
  for (int st = 128; st > 0; st >>= 1) {
    if (threadIdx.x < st) red[threadIdx.x] += red[threadIdx.x + st];
    __syncthreads();
  }
  if (threadIdx.x == 0) d[bco] = rsqrtf(red[0] + 1e-4f);
}

// ---------------------------------------------------------------- weight pack
// Pack modulated weights to bf16 in WMMA B-fragment layout:
// fragId = ((b*9+tap)*(CIN/32)+ciBlk)*(COUT/16)+coT ; per-lane K sequential.
__global__ void pack_kernel(const float* __restrict__ bw,
                            const float* __restrict__ s,
                            const float* __restrict__ d,
                            ushort_t* __restrict__ pw) {
  size_t o = (size_t)blockIdx.x * blockDim.x + threadIdx.x;
  const size_t total = (size_t)B_ * 9 * CIN_ * COUT_;   // 18,874,368
  if (o >= total) return;
  int e     = (int)(o & 15);
  int lane  = (int)((o >> 4)  & 31);
  int coT   = (int)((o >> 9)  & 31);     // COUT/16
  int ciBlk = (int)((o >> 14) & 15);     // CIN/32
  int tap   = (int)((o >> 18) % 9);
  int b     = (int)(o / ((size_t)9 << 18));
  int k  = ((lane >> 4) << 4) + e;       // B-frag: lane<16 -> K=e ; else K=16+e
  int ci = ciBlk * 32 + k;
  int co = coT * 16 + (lane & 15);
  const float cs = 1.0f / 1536.0f;
  float w = bw[((size_t)co * CIN_ + ci) * 9 + tap]
            * s[(size_t)b * CIN_ + ci] * cs * d[(size_t)b * COUT_ + co];
  pw[o] = f2bf(w);
}

// ---------------------------------------------------------------- WMMA conv3x3
// Block: 64 pixels (8x8 tile) x 128 couts. 8 waves: 4 M-groups x 2 N-groups.
// Wave: 16 px x 64 co -> 4 f32 accumulators; K = CIN*9 via 32-ch blocks x 9 taps.
// Register double-buffering overlaps next channel-block's global loads with WMMA.
template<int H, bool BF16IN>
__global__ __launch_bounds__(256)
void conv_wmma_kernel(const void* __restrict__ inv,     // [B][CIN][H][H] f32|bf16
                      const ushort_t* __restrict__ pw,  // packed B fragments
                      const float* __restrict__ bias,   // [COUT]
                      const float* __restrict__ noise,  // [B][H][H]
                      const float* __restrict__ nscale, // [1]
                      float* __restrict__ out) {        // [B][COUT][H][H]
  constexpr int W = H;
  constexpr int TILES_X = W / 8;
  constexpr int NSTG = 13;                              // ceil(3200/256)
  __shared__ __align__(16) float ldsF[128 * 64];        // 32 KB (epilogue)
  ushort_t* ldsU = (ushort_t*)ldsF;                     // staging [100 px][32 ch]

  const float*    inF = (const float*)inv;
  const ushort_t* inU = (const ushort_t*)inv;

  const int tid   = threadIdx.x;
  const int lane  = tid & 31;
  const int wv    = tid >> 5;
  const int waveM = wv & 3;            // 4 M-groups of 16 pixels
  const int waveN = wv >> 2;           // 2 N-groups of 64 couts
  const int tileX = blockIdx.x % TILES_X;
  const int tileY = blockIdx.x / TILES_X;
  const int coBlk = blockIdx.y;        // 0..3  (128 couts each)
  const int b     = blockIdx.z;

  const int m  = lane & 15;
  const int hi = lane >> 4;

  v8f acc[4];
  #pragma unroll
  for (int t = 0; t < 4; ++t) acc[t] = (v8f){0,0,0,0,0,0,0,0};

  const int y0 = tileY * 8 - 1;
  const int x0 = tileX * 8 - 1;
  const size_t inB = (size_t)b * CIN_ * H * W;
  const int coTile0 = coBlk * 8 + waveN * 4;

  // ---- precompute staging addresses (invariant across channel blocks) ----
  unsigned baseOff[NSTG];              // bit31 set => no global load (OOB/idle)
  int      lidx[NSTG];                 // -1 => no LDS store (idle)
  #pragma unroll
  for (int t = 0; t < NSTG; ++t) {
    int idx = tid + 256 * t;
    if (idx < 3200) {
      int c = idx / 100, p = idx % 100;
      int py = p / 10, px = p % 10;
      int gy = y0 + py, gx = x0 + px;
      lidx[t] = p * 32 + c;
      baseOff[t] = (gy >= 0 && gy < H && gx >= 0 && gx < W)
                       ? (unsigned)((c * H + gy) * W + gx)
                       : 0x80000000u;
    } else { lidx[t] = -1; baseOff[t] = 0x80000000u; }
  }

  ushort_t stg[NSTG];
  auto load_stage = [&](int cb) {
    #pragma unroll
    for (int t = 0; t < NSTG; ++t) {
      unsigned bo = baseOff[t];
      ushort_t v = 0;
      if (!(bo >> 31)) {
        size_t a = inB + (size_t)cb * (H * W) + bo;
        v = BF16IN ? inU[a] : f2bf(inF[a]);
      }
      stg[t] = v;
    }
  };
  auto store_stage = [&]() {
    #pragma unroll
    for (int t = 0; t < NSTG; ++t)
      if (lidx[t] >= 0) ldsU[lidx[t]] = stg[t];
  };

  load_stage(0);

  #pragma unroll 1
  for (int ciBlk = 0; ciBlk < CIN_ / 32; ++ciBlk) {
    store_stage();
    __syncthreads();
    if (ciBlk + 1 < CIN_ / 32) load_stage((ciBlk + 1) * 32);  // overlap w/ WMMA

    #pragma unroll
    for (int tap = 0; tap < 9; ++tap) {
      const int dy = tap / 3, dx = tap % 3;
      // A fragment: lane<16 holds K 0..15 ; lane>=16 holds 16..31
      const int pix = (2 * waveM + (m >> 3) + dy) * 10 + (m & 7) + dx;
      union { uint4 q[2]; v16bf v; } A;
      const ushort_t* row = ldsU + pix * 32;
      A.q[0] = *(const uint4*)(row + 16 * hi);         // K = 16*hi .. 16*hi+7
      A.q[1] = *(const uint4*)(row + 16 * hi + 8);     // K = 16*hi+8 .. +7

      const size_t fragBase =
          (((size_t)(b * 9 + tap) * (CIN_ / 32) + ciBlk) * 32 + coTile0) * 512
          + (size_t)lane * 16;
      #pragma unroll
      for (int t = 0; t < 4; ++t) {
        union { uint4 q[2]; v16bf v; } Bm;
        const uint4* bp = (const uint4*)(pw + fragBase + (size_t)t * 512);
        Bm.q[0] = bp[0]; Bm.q[1] = bp[1];
        acc[t] = __builtin_amdgcn_wmma_f32_16x16x32_bf16(
                     false, A.v, false, Bm.v, (short)0, acc[t], false, false);
      }
    }
    __syncthreads();
  }

  // epilogue: transpose via LDS -> coalesced float4 NCHW stores; fuse
  // bias + noise + leaky_relu(0.2)*sqrt(2)
  const int n = lane & 15;
  #pragma unroll
  for (int t = 0; t < 4; ++t)
    #pragma unroll
    for (int r = 0; r < 8; ++r) {
      int mm = r + 8 * hi;             // D layout: M = r + 8*(lane/16), N = lane%16
      int p2 = 16 * waveM + mm;
      ldsF[(waveN * 64 + t * 16 + n) * 64 + p2] = acc[t][r];
    }
  __syncthreads();

  const float ns = nscale[0];
  for (int j = tid; j < 2048; j += 256) {              // 8192 floats as float4
    int co_l = j >> 4;
    int p4   = (j & 15) << 2;                          // 4-aligned pixel in tile
    int pr = p4 >> 3, pc = p4 & 7;
    int oy = tileY * 8 + pr, ox = tileX * 8 + pc;
    int co = coBlk * 128 + co_l;
    float bco = bias[co];
    float4 v  = *(const float4*)&ldsF[co_l * 64 + p4];
    float4 nz = *(const float4*)&noise[((size_t)b * H + oy) * W + ox];
    float r[4] = {v.x + bco + nz.x * ns, v.y + bco + nz.y * ns,
                  v.z + bco + nz.z * ns, v.w + bco + nz.w * ns};
    #pragma unroll
    for (int q = 0; q < 4; ++q)
      r[q] = (r[q] > 0.f ? r[q] : 0.2f * r[q]) * SQRT2_;
    *(float4*)&out[(((size_t)b * COUT_ + co) * H + oy) * W + ox] =
        make_float4(r[0], r[1], r[2], r[3]);
  }
}

// ------------------------------------------- fused bilinear up x2 + 4x4 blur
// h1 (64 MB) is fully L2-resident; emit bf16 directly for conv2's staging.
__global__ void upblur_kernel(const float* __restrict__ h,   // [B][C][64][64]
                              ushort_t* __restrict__ o) {    // [B][C][128][128] bf16
  size_t g = (size_t)blockIdx.x * blockDim.x + threadIdx.x;
  if (g >= (size_t)B_ * COUT_ * 128 * 128) return;
  int ox = (int)(g & 127), oy = (int)((g >> 7) & 127);
  int c = (int)((g >> 14) & 511), b = (int)(g >> 23);
  const float* hp = h + ((size_t)b * COUT_ + c) * 64 * 64;
  const float F[4] = {1.f, 3.f, 3.f, 1.f};
  const float sc = 63.0f / 127.0f;     // align_corners: (n_in-1)/(n_out-1)
  float acc = 0.f;
  #pragma unroll
  for (int j = 0; j < 4; ++j) {
    int uy = oy - 2 + j;
    if (uy < 0 || uy > 127) continue;
    float cy = uy * sc;
    int yl = (int)cy; if (yl > 62) yl = 62;
    float wy = cy - (float)yl;
    float wy0 = F[j] * (1.f - wy), wy1 = F[j] * wy;
    const float* r0 = hp + yl * 64;
    const float* r1 = r0 + 64;
    #pragma unroll
    for (int i = 0; i < 4; ++i) {
      int ux = ox - 2 + i;
      if (ux < 0 || ux > 127) continue;
      float cx = ux * sc;
      int xl = (int)cx; if (xl > 62) xl = 62;
      float wx = cx - (float)xl;
      float wx0 = F[i] * (1.f - wx), wx1 = F[i] * wx;
      acc += wy0 * (wx0 * r0[xl] + wx1 * r0[xl + 1])
           + wy1 * (wx0 * r1[xl] + wx1 * r1[xl + 1]);
    }
  }
  o[g] = f2bf(acc * (1.0f / 64.0f));
}

// ---------------------------------------------------------------------- driver
extern "C" void kernel_launch(void* const* d_in, const int* in_sizes, int n_in,
                              void* d_out, int out_size, void* d_ws, size_t ws_size,
                              hipStream_t stream) {
  const float* x       = (const float*)d_in[0];
  const float* style0  = (const float*)d_in[1];
  const float* style1  = (const float*)d_in[2];
  const float* noise1  = (const float*)d_in[3];
  const float* noise2  = (const float*)d_in[4];
  const float* aff1_w  = (const float*)d_in[5];
  const float* aff1_b  = (const float*)d_in[6];
  const float* conv1_w = (const float*)d_in[7];
  const float* conv1_b = (const float*)d_in[8];
  const float* ns1     = (const float*)d_in[9];
  const float* aff2_w  = (const float*)d_in[10];
  const float* aff2_b  = (const float*)d_in[11];
  const float* conv2_w = (const float*)d_in[12];
  const float* conv2_b = (const float*)d_in[13];
  const float* ns2     = (const float*)d_in[14];

  char* ws = (char*)d_ws;
  const size_t PWB = (size_t)B_ * 9 * CIN_ * COUT_ * 2;     // 37,748,736 B
  float*    s1  = (float*)(ws);
  float*    s2  = (float*)(ws + 16384);
  float*    dd1 = (float*)(ws + 32768);
  float*    dd2 = (float*)(ws + 49152);
  ushort_t* pw1 = (ushort_t*)(ws + 65536);
  ushort_t* pw2 = (ushort_t*)(ws + 65536 + PWB);
  float*    h1  = (float*)(ws + 65536 + 2 * PWB);                     // 64 MB f32
  ushort_t* hb  = (ushort_t*)(ws + 65536 + 2 * PWB + (size_t)67108864); // 128 MB bf16

  style_kernel<<<16, 256, 0, stream>>>(style0, aff1_w, aff1_b, s1);
  style_kernel<<<16, 256, 0, stream>>>(style1, aff2_w, aff2_b, s2);
  demod_kernel<<<B_ * COUT_, 256, 0, stream>>>(conv1_w, s1, dd1);
  demod_kernel<<<B_ * COUT_, 256, 0, stream>>>(conv2_w, s2, dd2);
  pack_kernel<<<73728, 256, 0, stream>>>(conv1_w, s1, dd1, pw1);
  pack_kernel<<<73728, 256, 0, stream>>>(conv2_w, s2, dd2, pw2);

  conv_wmma_kernel<64, false><<<dim3(64, 4, 8), 256, 0, stream>>>(
      x, pw1, conv1_b, noise1, ns1, h1);
  upblur_kernel<<<262144, 256, 0, stream>>>(h1, hb);
  conv_wmma_kernel<128, true><<<dim3(256, 4, 8), 256, 0, stream>>>(
      hb, pw2, conv2_b, noise2, ns2, (float*)d_out);
}
